// CustomCrossAttention_21406117003981
// MI455X (gfx1250) — compile-verified, matrix operand order checked by reading, not run on hardware
//
#include <hip/hip_runtime.h>
#include <hip/hip_bf16.h>
#include <stdint.h>

// ---------------------------------------------------------------------------
// CDNA5 / gfx1250: wave32, WMMA 16x16x32 bf16 (f32 acc) + TDM tensor loads.
// ---------------------------------------------------------------------------

typedef __bf16 bf16_t;
typedef __attribute__((ext_vector_type(16))) __bf16 bf16x16;
typedef __attribute__((ext_vector_type(8)))  __bf16 bf16x8;
typedef __attribute__((ext_vector_type(8)))  float  f32x8;
typedef __attribute__((ext_vector_type(4)))  unsigned int u32x4;
typedef __attribute__((ext_vector_type(8)))  int i32x8;
typedef __attribute__((ext_vector_type(4)))  int i32x4;

#define BB 16
#define TT 2048
#define DD 256

__device__ __forceinline__ f32x8 wmma_bf16(bf16x16 a, bf16x16 b, f32x8 c) {
  // D = A(16x32) x B(32x16) + C(16x16 f32)
  return __builtin_amdgcn_wmma_f32_16x16x32_bf16(false, a, false, b, (short)0, c,
                                                 false, false);
}

// Load a 16x32 bf16 fragment in the ISA A-layout from a row-major [rows x K]
// bf16 source. Also valid for the B operand when the source is the row-major
// TRANSPOSE of B (i.e. [N x K]): lane -> row = lane&15, half h = lane>>4,
// elems 0..7  -> K = k0 + h*8 + e
// elems 8..15 -> K = k0 + 16 + h*8 + (e-8)
__device__ __forceinline__ bf16x16 load_frag_bf16(const bf16_t* base, int stride,
                                                  int row, int k0) {
  int lane = threadIdx.x & 31;
  int m = lane & 15, h = lane >> 4;
  const bf16_t* p = base + (size_t)(row + m) * stride + k0 + h * 8;
  bf16x8 lo = *(const bf16x8*)p;
  bf16x8 hi = *(const bf16x8*)(p + 16);
  bf16x16 r;
#pragma unroll
  for (int i = 0; i < 8; ++i) { r[i] = lo[i]; r[i + 8] = hi[i]; }
  return r;
}

// Same fragment layout, sourced from row-major fp32 (converted to bf16).
__device__ __forceinline__ bf16x16 load_frag_f32(const float* base, int stride,
                                                 int row, int k0) {
  int lane = threadIdx.x & 31;
  int m = lane & 15, h = lane >> 4;
  const float* p = base + (size_t)(row + m) * stride + k0 + h * 8;
  bf16x16 r;
#pragma unroll
  for (int i = 0; i < 8; ++i) {
    r[i]     = (bf16_t)p[i];
    r[i + 8] = (bf16_t)p[16 + i];
  }
  return r;
}

// ---------------------------------------------------------------------------
// TDM: issue a 2D bf16 tile load (tileW x tileH of a tensor with row stride
// `strideElems`, tensor dims dimW x dimH) from global memory into LDS.
// D# built per CDNA5 ISA 8.3/8.4.  clang-23 6-arg builtin form.
// ---------------------------------------------------------------------------
#if __has_builtin(__builtin_amdgcn_tensor_load_to_lds)
#define HAVE_TDM 1
__device__ __forceinline__ void tdm_load_tile_bf16(const bf16_t* gsrc, void* lds_dst,
                                                   int tileW, int tileH,
                                                   int dimW, int dimH,
                                                   int strideElems) {
  unsigned lds_off = (unsigned)(uintptr_t)lds_dst;  // LDS aperture: addr[31:0]
  unsigned long long ga = (unsigned long long)(uintptr_t)gsrc;
  u32x4 g0;
  g0[0] = 1u;                                        // count=1 (valid), user mode
  g0[1] = lds_off;                                   // [63:32]  lds_addr
  g0[2] = (unsigned)(ga & 0xFFFFFFFFu);              // [95:64]  global_addr lo
  g0[3] = (unsigned)((ga >> 32) & 0x01FFFFFFu)       // [120:96] global_addr hi
        | (2u << 30);                                // [127:126] type = 2
  i32x8 g1;
  g1[0] = (int)(1u << 16);                           // data_size = 1 (2 bytes)
  g1[1] = (int)(((unsigned)dimW & 0xFFFFu) << 16);   // tensor_dim0[15:0] @ [63:48]
  g1[2] = (int)((((unsigned)dimW >> 16) & 0xFFFFu))  // tensor_dim0[31:16]
        | (int)(((unsigned)dimH & 0xFFFFu) << 16);   // tensor_dim1[15:0] @ [95:80]
  g1[3] = (int)((((unsigned)dimH >> 16) & 0xFFFFu))  // tensor_dim1[31:16]
        | (int)(((unsigned)tileW & 0xFFFFu) << 16);  // tile_dim0 @ [127:112]
  g1[4] = (int)((unsigned)tileH & 0xFFFFu);          // tile_dim1 @ [143:128], tile_dim2=0
  g1[5] = strideElems;                               // tensor_dim0_stride[31:0]
  g1[6] = 0;                                         // stride0 hi, dim1_stride lo
  g1[7] = 0;
  i32x4 gz4 = {};                                    // 2D tensor: groups 2/3 unused
  i32x8 gz8 = {};
  __builtin_amdgcn_tensor_load_to_lds(g0, g1, gz4, gz4, gz8, 0);
}
#else
#define HAVE_TDM 0
#endif

// ---------------------------------------------------------------------------
// 1) per-batch feature means
// ---------------------------------------------------------------------------
__global__ void k_mean(const float* __restrict__ X, float* __restrict__ mu) {
  int b = blockIdx.x, j = threadIdx.x;
  const float* p = X + (size_t)b * TT * DD + j;
  float s = 0.f;
  for (int t = 0; t < TT; ++t) s += p[(size_t)t * DD];
  mu[b * DD + j] = s * (1.0f / TT);
}

// ---------------------------------------------------------------------------
// 2) covariance: Sigma = X^T X/(T-1) - T/(T-1) mu mu^T + 1e-6 I  (WMMA)
// ---------------------------------------------------------------------------
__global__ void k_gram(const float* __restrict__ X, const float* __restrict__ mu,
                       float* __restrict__ Sig) {
  __shared__ __attribute__((aligned(16))) bf16_t Xi[64 * 32];  // [feat][t]
  __shared__ __attribute__((aligned(16))) bf16_t Xj[64 * 32];  // [feat][t]
  int bx = blockIdx.x;
  int tj = bx & 3, ti = (bx >> 2) & 3, b = bx >> 4;
  int tid = threadIdx.x;
  int w = tid >> 5, lane = tid & 31, h = lane >> 4, n = lane & 15;
  int ri = (w & 3) * 16;   // row sub-block within 64
  int cj = (w >> 2) * 32;  // col sub-block within 64
  f32x8 acc0 = {}, acc1 = {};
  for (int tc = 0; tc < TT; tc += 32) {
    __syncthreads();
#pragma unroll
    for (int i = 0; i < 8; ++i) {
      int e = tid + 256 * i;          // 0..2047
      int t = e >> 6, f = e & 63;
      const float* src = X + ((size_t)b * TT + tc + t) * DD;
      Xi[f * 32 + t] = (bf16_t)src[ti * 64 + f];
      Xj[f * 32 + t] = (bf16_t)src[tj * 64 + f];
    }
    __syncthreads();
    bf16x16 a  = load_frag_bf16(Xi, 32, ri, 0);
    bf16x16 b0 = load_frag_bf16(Xj, 32, cj, 0);
    bf16x16 b1 = load_frag_bf16(Xj, 32, cj + 16, 0);
    acc0 = wmma_bf16(a, b0, acc0);
    acc1 = wmma_bf16(a, b1, acc1);
  }
  const float invTm1 = 1.0f / (float)(TT - 1);
  const float tf = (float)TT * invTm1;
#pragma unroll
  for (int r = 0; r < 8; ++r) {
    int i  = ti * 64 + ri + r + 8 * h;
    int j0 = tj * 64 + cj + n;
    int j1 = j0 + 16;
    float mi = mu[b * DD + i];
    float g0 = acc0[r] * invTm1 - tf * mi * mu[b * DD + j0];
    float g1 = acc1[r] * invTm1 - tf * mi * mu[b * DD + j1];
    if (i == j0) g0 += 1e-6f;
    if (i == j1) g1 += 1e-6f;
    Sig[(size_t)b * DD * DD + (size_t)i * DD + j0] = g0;
    Sig[(size_t)b * DD * DD + (size_t)i * DD + j1] = g1;
  }
}

// ---------------------------------------------------------------------------
// 3) Cholesky, whole 256x256 matrix in 256 KB dynamic LDS (320 KB WGP pool).
//    Emits L as bf16 row-major [i][k] (upper = 0) = B^T layout for resample.
// ---------------------------------------------------------------------------
__global__ void k_chol(const float* __restrict__ Sig, bf16_t* __restrict__ Lb) {
  extern __shared__ float S[];
  int b = blockIdx.x, tid = threadIdx.x;
  for (int e = tid; e < DD * DD; e += 256) S[e] = Sig[(size_t)b * DD * DD + e];
  __syncthreads();
  for (int k = 0; k < DD; ++k) {
    if (tid == 0) S[k * DD + k] = sqrtf(fmaxf(S[k * DD + k], 1e-12f));
    __syncthreads();
    float dk = S[k * DD + k];
    int i = k + 1 + tid;
    if (i < DD) S[i * DD + k] /= dk;
    __syncthreads();
    if (i < DD) {
      float lik = S[i * DD + k];
      for (int j = k + 1; j <= i; ++j) S[i * DD + j] -= lik * S[j * DD + k];
    }
    __syncthreads();
  }
  for (int e = tid; e < DD * DD; e += 256) {
    int i = e >> 8, j = e & 255;
    Lb[(size_t)b * DD * DD + e] = (bf16_t)((j <= i) ? S[e] : 0.0f);
  }
}

// ---------------------------------------------------------------------------
// 4) K = mu + eps @ L^T ; eps generated in-register in WMMA A-fragment layout.
// ---------------------------------------------------------------------------
__device__ __forceinline__ float rnorm_hash(unsigned seed, unsigned idx) {
  unsigned x = (idx ^ seed) * 0x9E3779B9u;
  x ^= x >> 16; x *= 0x7FEB352Du; x ^= x >> 15; x *= 0x846CA68Bu; x ^= x >> 16;
  unsigned y = (idx + 0x6D2B79F5u) * (seed | 1u);
  y ^= y >> 13; y *= 0xC2B2AE35u; y ^= y >> 16;
  float u1 = fmaxf((float)(x >> 8) * (1.0f / 16777216.0f), 1e-7f);
  float u2 = (float)(y >> 8) * (1.0f / 16777216.0f);
  return sqrtf(-2.0f * __logf(u1)) * __cosf(6.28318530718f * u2);
}

__global__ void k_resample(const float* __restrict__ mu, const bf16_t* __restrict__ Lb,
                           bf16_t* __restrict__ Kout, unsigned seed) {
  int bx = blockIdx.x;
  int b = bx >> 5, tb = bx & 31;
  int tid = threadIdx.x, w = tid >> 5, lane = tid & 31, h = lane >> 4, n = lane & 15;
  int t0 = tb * 64 + (w & 3) * 16;   // this wave's 16 sample rows
  int cb = (w >> 2) * 128;           // this wave's 128 output features
  const bf16_t* L = Lb + (size_t)b * DD * DD;
  f32x8 acc[8] = {};
  int trow = t0 + n;                 // A-operand row for this lane
  unsigned base_idx = (unsigned)((b * TT + trow) * DD);
  for (int k0 = 0; k0 < DD; k0 += 32) {
    bf16x16 ea;
#pragma unroll
    for (int e = 0; e < 16; ++e) {
      int k = k0 + h * 8 + e + ((e >= 8) ? 8 : 0);
      ea[e] = (bf16_t)rnorm_hash(seed, base_idx + (unsigned)k);
    }
#pragma unroll
    for (int nf = 0; nf < 8; ++nf) {
      bf16x16 bl = load_frag_bf16(L, DD, cb + nf * 16, k0);
      acc[nf] = wmma_bf16(ea, bl, acc[nf]);
    }
  }
#pragma unroll
  for (int nf = 0; nf < 8; ++nf) {
#pragma unroll
    for (int r = 0; r < 8; ++r) {
      int t = t0 + r + 8 * h;
      int c = cb + nf * 16 + n;
      Kout[((size_t)b * TT + t) * DD + c] = (bf16_t)(acc[nf][r] + mu[b * DD + c]);
    }
  }
}

// ---------------------------------------------------------------------------
// 5) Flash attention: Out = softmax(Q K^T / 16) @ Val.
//    K-tile: TDM tensor_load_to_lds (wave 0 issues, s_wait_tensorcnt).
//    V-tile: manual fp32->bf16 transpose into LDS (overlaps the DMA).
// ---------------------------------------------------------------------------
__global__ void k_attn(const float* __restrict__ Q, const bf16_t* __restrict__ Kbf,
                       const float* __restrict__ Val, bf16_t* __restrict__ Out) {
  __shared__ __attribute__((aligned(16))) bf16_t Kt[32 * DD];    // [key][feat]
  __shared__ __attribute__((aligned(16))) bf16_t Vt[DD * 32];    // [feat][key]
  __shared__ __attribute__((aligned(16))) bf16_t Pb[8][16 * 32]; // per-wave P
  int bx = blockIdx.x;
  int b = bx >> 5, tb = bx & 31;
  int tid = threadIdx.x, w = tid >> 5, lane = tid & 31, h = lane >> 4, n = lane & 15;
  int t0 = tb * 64 + (w & 3) * 16;
  int fb = (w >> 2) * 128;
  const float* qbase = Q + ((size_t)b * TT + t0) * DD;
  bf16x16 qf[8];
#pragma unroll
  for (int kc = 0; kc < 8; ++kc) qf[kc] = load_frag_f32(qbase, DD, 0, kc * 32);
  f32x8 acc[8] = {};
  float mrow[8], lrow[8];
#pragma unroll
  for (int r = 0; r < 8; ++r) { mrow[r] = -1e30f; lrow[r] = 0.f; }
  bf16_t* Pw = &Pb[w][0];
  for (int s0 = 0; s0 < TT; s0 += 32) {
    __syncthreads();
    const bf16_t* ksrc = Kbf + ((size_t)b * TT + s0) * DD;
#if HAVE_TDM
    if (tid < 32) {  // wave 0 kicks the K-tile DMA: 256x32 bf16, stride 256
      tdm_load_tile_bf16(ksrc, &Kt[0], DD, 32, DD, TT - s0, DD);
    }
#else
    {
      const uint32_t* ks = (const uint32_t*)ksrc;
      uint32_t* kd = (uint32_t*)Kt;
#pragma unroll
      for (int i = 0; i < 16; ++i) kd[tid + 256 * i] = ks[tid + 256 * i];
    }
#endif
    const float* vs = Val + ((size_t)b * TT + s0) * DD;
    if (s0 + 32 < TT) __builtin_prefetch(vs + (size_t)32 * DD, 0, 1);
#pragma unroll
    for (int i = 0; i < 32; ++i) {
      int e = tid + 256 * i;
      int t = e >> 8, f = e & 255;
      Vt[f * 32 + t] = (bf16_t)vs[(size_t)t * DD + f];
    }
#if HAVE_TDM
    if (tid < 32) __builtin_amdgcn_s_wait_tensorcnt(0);
#endif
    __syncthreads();
    // scores: 16 queries x 32 keys, K = 256 features
    f32x8 sc0 = {}, sc1 = {};
#pragma unroll
    for (int kc = 0; kc < 8; ++kc) {
      bf16x16 b0 = load_frag_bf16(Kt, DD, 0, kc * 32);
      bf16x16 b1 = load_frag_bf16(Kt, DD, 16, kc * 32);
      sc0 = wmma_bf16(qf[kc], b0, sc0);
      sc1 = wmma_bf16(qf[kc], b1, sc1);
    }
    // online softmax (rows M = r + 8h live in 16-lane halves)
    float corr[8];
#pragma unroll
    for (int r = 0; r < 8; ++r) {
      float v0 = sc0[r] * 0.0625f;
      float v1 = sc1[r] * 0.0625f;
      float mx = fmaxf(v0, v1);
      mx = fmaxf(mx, __shfl_xor(mx, 1));
      mx = fmaxf(mx, __shfl_xor(mx, 2));
      mx = fmaxf(mx, __shfl_xor(mx, 4));
      mx = fmaxf(mx, __shfl_xor(mx, 8));
      float mn = fmaxf(mrow[r], mx);
      float c = __expf(mrow[r] - mn);
      mrow[r] = mn;
      float p0 = __expf(v0 - mn);
      float p1 = __expf(v1 - mn);
      float rs = p0 + p1;
      rs += __shfl_xor(rs, 1);
      rs += __shfl_xor(rs, 2);
      rs += __shfl_xor(rs, 4);
      rs += __shfl_xor(rs, 8);
      lrow[r] = lrow[r] * c + rs;
      corr[r] = c;
      int pr = r + 8 * h;                 // C-layout -> LDS row-major P
      Pw[pr * 32 + n]      = (bf16_t)p0;
      Pw[pr * 32 + 16 + n] = (bf16_t)p1;
    }
#pragma unroll
    for (int nf = 0; nf < 8; ++nf)
#pragma unroll
      for (int r = 0; r < 8; ++r) acc[nf][r] *= corr[r];
    // att += P(16x32) @ V(32 x 128-feature slice)
    bf16x16 pa = load_frag_bf16(Pw, 32, 0, 0);
#pragma unroll
    for (int nf = 0; nf < 8; ++nf) {
      bf16x16 bv = load_frag_bf16(Vt, 32, fb + nf * 16, 0);
      acc[nf] = wmma_bf16(pa, bv, acc[nf]);
    }
  }
#pragma unroll
  for (int nf = 0; nf < 8; ++nf) {
#pragma unroll
    for (int r = 0; r < 8; ++r) {
      int t = t0 + r + 8 * h;
      int c = fb + nf * 16 + n;
      Out[((size_t)b * TT + t) * DD + c] = (bf16_t)(acc[nf][r] / lrow[r]);
    }
  }
}

// ---------------------------------------------------------------------------
// 6) gated fusion + cosine blend; one wave per (b,t) row.
// ---------------------------------------------------------------------------
__global__ void k_fuse(const float* __restrict__ A, const float* __restrict__ V,
                       const bf16_t* __restrict__ Aav, const bf16_t* __restrict__ Ava,
                       const float* __restrict__ WA, const float* __restrict__ WV,
                       const float* __restrict__ bA, const float* __restrict__ bV,
                       float* __restrict__ out) {
  int row = blockIdx.x * 8 + (threadIdx.x >> 5);
  int lane = threadIdx.x & 31;
  size_t base = (size_t)row * DD;
  int f0 = lane * 8;
  float a[8], v[8], aav[8], ava[8];
#pragma unroll
  for (int i = 0; i < 8; ++i) {
    a[i]   = A[base + f0 + i];
    v[i]   = V[base + f0 + i];
    aav[i] = (float)Aav[base + f0 + i];
    ava[i] = (float)Ava[base + f0 + i];
  }
  float dot = 0.f, na = 0.f, nv = 0.f, ga = 0.f, gv = 0.f;
#pragma unroll
  for (int i = 0; i < 8; ++i) {
    dot += a[i] * v[i];
    na  += a[i] * a[i];
    nv  += v[i] * v[i];
    ga  += a[i] * WA[f0 + i] + aav[i] * WA[DD + f0 + i];
    gv  += v[i] * WV[f0 + i] + ava[i] * WV[DD + f0 + i];
  }
#pragma unroll
  for (int m = 1; m < 32; m <<= 1) {
    dot += __shfl_xor(dot, m);
    na  += __shfl_xor(na, m);
    nv  += __shfl_xor(nv, m);
    ga  += __shfl_xor(ga, m);
    gv  += __shfl_xor(gv, m);
  }
  float gA = 1.f / (1.f + __expf(-(ga + bA[0])));
  float gV = 1.f / (1.f + __expf(-(gv + bV[0])));
  float cosv  = dot / fmaxf(sqrtf(na) * sqrtf(nv), 1e-8f);
  float alpha = 1.f / (1.f + __expf(-cosv));
  float beta  = 1.f - alpha;
#pragma unroll
  for (int i = 0; i < 8; ++i) {
    float fa = gA * a[i] + (1.f - gA) * aav[i];
    float fv = gV * v[i] + (1.f - gV) * ava[i];
    out[base + f0 + i] = alpha * fa + beta * fv;
  }
}

// ---------------------------------------------------------------------------
// launch: V-stats -> Kv ; A-stats -> Ka ; attn x2 ; fuse
// workspace layout (bytes, 16B aligned), ~70 MB total:
//   mu      @ 0         (16 KB, reused per modality)
//   Sigma   @ 16384     (4 MB f32, reused)
//   L bf16  @ 4210688   (2 MB, reused)
//   Kv bf16 @ 6307840   (16 MB)
//   Ka bf16 @ 23085056  (16 MB)
//   att_av  @ 39862272  (16 MB bf16)
//   att_va  @ 56639488  (16 MB bf16)
// ---------------------------------------------------------------------------
extern "C" void kernel_launch(void* const* d_in, const int* in_sizes, int n_in,
                              void* d_out, int out_size, void* d_ws, size_t ws_size,
                              hipStream_t stream) {
  const float* A  = (const float*)d_in[0];
  const float* V  = (const float*)d_in[1];
  const float* WA = (const float*)d_in[2];
  const float* WV = (const float*)d_in[3];
  const float* bA = (const float*)d_in[4];
  const float* bV = (const float*)d_in[5];
  float* out = (float*)d_out;
  char* ws = (char*)d_ws;
  float*  mu  = (float*)(ws + 0);
  float*  Sig = (float*)(ws + 16384);
  bf16_t* Lb  = (bf16_t*)(ws + 4210688);
  bf16_t* Kv  = (bf16_t*)(ws + 6307840);
  bf16_t* Ka  = (bf16_t*)(ws + 23085056);
  bf16_t* Aav = (bf16_t*)(ws + 39862272);
  bf16_t* Ava = (bf16_t*)(ws + 56639488);

  // Visual modality stats -> K_v (seed 42)
  k_mean    <<<BB,      256, 0, stream>>>(V, mu);
  k_gram    <<<BB * 16, 256, 0, stream>>>(V, mu, Sig);
  k_chol    <<<BB,      256, DD * DD * sizeof(float), stream>>>(Sig, Lb);
  k_resample<<<BB * 32, 256, 0, stream>>>(mu, Lb, Kv, 42u);
  // Audio modality stats -> K_a (seed 43)
  k_mean    <<<BB,      256, 0, stream>>>(A, mu);
  k_gram    <<<BB * 16, 256, 0, stream>>>(A, mu, Sig);
  k_chol    <<<BB,      256, DD * DD * sizeof(float), stream>>>(Sig, Lb);
  k_resample<<<BB * 32, 256, 0, stream>>>(mu, Lb, Ka, 43u);
  // Cross attention (both directions)
  k_attn<<<BB * 32, 256, 0, stream>>>(A, Kv, V, Aav);
  k_attn<<<BB * 32, 256, 0, stream>>>(V, Ka, A, Ava);
  // Gated fusion + cosine blend
  k_fuse<<<BB * TT / 8, 256, 0, stream>>>(A, V, Aav, Ava, WA, WV, bA, bV, out);
}
